// Gudi_UpProj_Block_42417097016471
// MI455X (gfx1250) — matrix operand order repaired and checked by reading, not compiled, not run on hardware
//
#include <hip/hip_runtime.h>
#include <cstdint>
#include <cstddef>

typedef __attribute__((ext_vector_type(4)))  unsigned int v4u;
typedef __attribute__((ext_vector_type(8)))  unsigned int v8u;
typedef __attribute__((ext_vector_type(16))) __bf16      v16bf;
typedef __attribute__((ext_vector_type(8)))  float       v8f;

#define BN_COUNT 100352.0f   /* 8*112*112 elements per channel */
#define BN_EPS   1e-5f

struct Taps { int n; int di[9]; int dj[9]; int woff[9]; };

__device__ __forceinline__ unsigned short f2bf(float f) {
  unsigned u = __builtin_bit_cast(unsigned, f);
  unsigned r = u + 0x7FFFu + ((u >> 16) & 1u);   // round-to-nearest-even
  return (unsigned short)(r >> 16);
}

// Async global->LDS copy of 16B per lane (CDNA5 GLOBAL_LOAD_ASYNC_TO_LDS_B128).
// vdst VGPR holds the LDS byte address (= low 32 bits of the generic pointer,
// per the LDS aperture mapping); vaddr is the 64-bit global
// address. Completion is tracked with ASYNCcnt.
__device__ __forceinline__ void async_ld_b128(const unsigned short* g, const unsigned short* lds) {
  unsigned loff = (unsigned)(size_t)lds;
  unsigned long long ga = (unsigned long long)(size_t)g;
  asm volatile("global_load_async_to_lds_b128 %0, %1, off"
               :: "v"(loff), "v"(ga) : "memory");
}
__device__ __forceinline__ void async_wait2() {
  asm volatile("s_wait_asynccnt 0x2" ::: "memory");
}
__device__ __forceinline__ void async_wait0() {
  asm volatile("s_wait_asynccnt 0x0" ::: "memory");
}

__device__ __forceinline__ v16bf mk_frag(v4u lo, v4u hi) {
  v8u t;
  t[0] = lo[0]; t[1] = lo[1]; t[2] = lo[2]; t[3] = lo[3];
  t[4] = hi[0]; t[5] = hi[1]; t[6] = hi[2]; t[7] = hi[3];
  return __builtin_bit_cast(v16bf, t);
}

// ---------------------------------------------------------------- utilities
__global__ void k_zero(float* p, int n) {
  int i = blockIdx.x * 256 + threadIdx.x;
  if (i < n) p[i] = 0.f;
}

// x: NCHW f32 (8,512,56,56) -> xp: NHWC bf16 [b][h][w][ci]
__global__ void k_pack_x(const float* __restrict__ x, unsigned short* __restrict__ xp, int n) {
  int e = blockIdx.x * 256 + threadIdx.x;
  if (e >= n) return;
  int ci = e & 511; int t = e >> 9;
  int w = t % 56; t /= 56;
  int h = t % 56;
  int b = t / 56;
  xp[e] = f2bf(x[((size_t)((b * 512 + ci) * 56 + h)) * 56 + w]);
}

// w: OIHW f32 (256,CinC,K,K) -> wp: [kh][kw][co][ci] bf16
__global__ void k_pack_w(const float* __restrict__ w, unsigned short* __restrict__ wp,
                         int CinC, int K, int n) {
  int e = blockIdx.x * 256 + threadIdx.x;
  if (e >= n) return;
  int ci = e % CinC; int t = e / CinC;
  int co = t & 255;  t >>= 8;
  int kw = t % K;
  int kh = t / K;
  wp[e] = f2bf(w[((size_t)((co * CinC + ci) * K + kh)) * K + kw]);
}

// ---------------------------------------------------------------- WMMA conv-GEMM
// Implicit GEMM: M = B*Hp*Wp output pixels, N = 256 output channels,
// K = taps.n * CinC, consumed as a flattened chunk stream (32 K per chunk).
// Block tile 128(M) x 128(N); 8 waves, each 32x64 (8 WMMA tiles).
// Software pipeline: A prefetched to regs (zero-filled at halo), B streamed
// global->LDS with async b128 copies, double-buffered LDS.
__global__ __launch_bounds__(256)
void k_conv_wmma(const unsigned short* __restrict__ inp,
                 const unsigned short* __restrict__ wgt,
                 float* __restrict__ outp,
                 float* __restrict__ gsum, float* __restrict__ gsumsq,
                 int Hin, int Win, int CinC,
                 int Hp, int Wp, int sh, int oh0, int sw, int ow0,
                 Taps taps)
{
  __shared__ unsigned short As[2][128 * 32];
  __shared__ unsigned short Bs[2][128 * 32];
  __shared__ float csum[128];
  __shared__ float csq[128];
  __shared__ int sdi[9], sdj[9], swoff[9];

  const int tid  = threadIdx.x;
  const int lane = tid & 31;
  const int wave = tid >> 5;
  const int wm   = wave >> 1;          // 0..3 : 32-row slab of M
  const int wn   = wave & 1;           // 0..1 : 64-col slab of N

  if (tid < 128) { csum[tid] = 0.f; csq[tid] = 0.f; }
  if (tid < 9) { sdi[tid] = taps.di[tid]; sdj[tid] = taps.dj[tid]; swoff[tid] = taps.woff[tid]; }
  __syncthreads();

  const int m0 = blockIdx.x * 128;
  const int n0 = blockIdx.y * 128;

  // Staging assignment (same shape for A and B tiles): 128 rows x 32 K,
  // thread -> (row 0..127, 16-element half).
  const int row  = tid >> 1;
  const int half = tid & 1;
  const int stOff = row * 32 + half * 16;

  const int mA = m0 + row;
  const int bA = mA / (Hp * Wp);
  const int rA = mA - bA * (Hp * Wp);
  const int iA = rA / Wp;
  const int jA = rA - iA * Wp;

  const int kchunks = CinC >> 5;
  const int kshift  = __popc(kchunks - 1);
  const int kmask   = kchunks - 1;
  const int total   = taps.n << kshift;

  const unsigned short* brow = wgt + (size_t)(n0 + row) * CinC + half * 16;

  // A prefetch into registers (conditional zero at halo)
  v4u a0, a1, a0n, a1n;
  auto issueA = [&](int c, v4u& r0, v4u& r1) {
    const int t  = c >> kshift;
    const int kc = c & kmask;
    const int ih = iA + sdi[t];
    const int jw = jA + sdj[t];
    const bool valid = (ih >= 0) & (ih < Hin) & (jw >= 0) & (jw < Win);
    const unsigned short* p =
        inp + (size_t)((bA * Hin + ih) * Win + jw) * CinC + kc * 32 + half * 16;
    v4u z = {0u, 0u, 0u, 0u};
    r0 = z; r1 = z;
    if (valid) { r0 = *(const v4u*)p; r1 = *(const v4u*)(p + 8); }
  };
  auto issueB = [&](int c, int buf) {
    const int t  = c >> kshift;
    const int kc = c & kmask;
    const unsigned short* p = brow + (size_t)swoff[t] + kc * 32;
    async_ld_b128(p,     &Bs[buf][stOff]);
    async_ld_b128(p + 8, &Bs[buf][stOff + 8]);
  };

  v8f acc[2][4];
#pragma unroll
  for (int am = 0; am < 2; ++am)
#pragma unroll
    for (int tn = 0; tn < 4; ++tn) acc[am][tn] = v8f{};

  // Fragment addressing per CDNA5 WMMA VGPR maps (ISA 7.12.2):
  //  A (16x32, 16-bit): lanes<16 hold K {0..7,16..23}; lanes>=16 hold {8..15,24..31}
  //  B (32x16, 16-bit): lanes<16 hold K {0..7,16..23}; lanes>=16 hold {8..15,24..31}
  const int fr  = lane & 15;
  const int ka0 = (lane < 16) ? 0 : 8;
  const int kb0 = (lane < 16) ? 0 : 8;

  issueA(0, a0, a1);
  issueB(0, 0);

  int cur = 0;
  for (int c = 0; c < total; ++c) {
    // stage prefetched A registers into the current LDS buffer
    *(v4u*)&As[cur][stOff]     = a0;
    *(v4u*)&As[cur][stOff + 8] = a1;

    const bool more = (c + 1 < total);
    if (more) {
      issueB(c + 1, cur ^ 1);       // async into the other buffer
      issueA(c + 1, a0n, a1n);      // global loads overlap compute below
    }
    if (more) async_wait2();        // only the older async pair must be done
    else      async_wait0();
    __syncthreads();

#pragma unroll
    for (int am = 0; am < 2; ++am) {
      const int arow = wm * 32 + am * 16 + fr;
      v4u alo = *(const v4u*)&As[cur][arow * 32 + ka0];
      v4u ahi = *(const v4u*)&As[cur][arow * 32 + ka0 + 16];
      v16bf af = mk_frag(alo, ahi);
#pragma unroll
      for (int tn = 0; tn < 4; ++tn) {
        const int col = wn * 64 + tn * 16 + fr;
        v4u blo = *(const v4u*)&Bs[cur][col * 32 + kb0];
        v4u bhi = *(const v4u*)&Bs[cur][col * 32 + kb0 + 16];
        v16bf bf = mk_frag(blo, bhi);
        acc[am][tn] = __builtin_amdgcn_wmma_f32_16x16x32_bf16(
            false, af, false, bf, (short)0, acc[am][tn], false, false);
      }
    }
    __syncthreads();
    a0 = a0n; a1 = a1n;
    cur ^= 1;
  }

  // Epilogue: store NHWC f32 + per-channel BN statistics (LDS reduce -> global).
  const int rowOff = (lane < 16) ? 0 : 8;   // C/D map: lanes>=16 -> M=r+8
#pragma unroll
  for (int am = 0; am < 2; ++am) {
#pragma unroll
    for (int tn = 0; tn < 4; ++tn) {
      const int colLocal = wn * 64 + tn * 16 + fr;
      const int co = n0 + colLocal;
      float s = 0.f, s2 = 0.f;
#pragma unroll
      for (int r = 0; r < 8; ++r) {
        float v = acc[am][tn][r];
        s  += v;
        s2 += v * v;
        const int m  = m0 + wm * 32 + am * 16 + rowOff + r;
        const int b  = m / (Hp * Wp);
        const int rm = m - b * (Hp * Wp);
        const int ii = rm / Wp;
        const int jj = rm - ii * Wp;
        const int oh = ii * sh + oh0;
        const int ow = jj * sw + ow0;
        outp[(size_t)((b * 112 + oh) * 112 + ow) * 256 + co] = v;
      }
      atomicAdd(&csum[colLocal], s);
      atomicAdd(&csq[colLocal], s2);
    }
  }
  __syncthreads();
  if (tid < 128) {
    atomicAdd(&gsum[n0 + tid],   csum[tid]);
    atomicAdd(&gsumsq[n0 + tid], csq[tid]);
  }
}

// ---------------------------------------------------------------- BN + relu + bf16 repack
__global__ void k_bn_relu_pack(const float* __restrict__ y,
                               const float* __restrict__ sum, const float* __restrict__ sumsq,
                               const float* __restrict__ gamma, const float* __restrict__ beta,
                               unsigned short* __restrict__ ap, int n) {
  int e = blockIdx.x * 256 + threadIdx.x;
  if (e >= n) return;
  int co = e & 255;
  float mean = sum[co] * (1.0f / BN_COUNT);
  float var  = sumsq[co] * (1.0f / BN_COUNT) - mean * mean;
  float inv  = rsqrtf(var + BN_EPS);
  float v = (y[e] - mean) * inv * gamma[co] + beta[co];
  v = v > 0.f ? v : 0.f;
  ap[e] = f2bf(v);
}

// ---------------------------------------------------------------- final BN + add + relu -> NCHW
__global__ void k_final(const float* __restrict__ y2, const float* __restrict__ ysc,
                        const float* __restrict__ s2, const float* __restrict__ q2,
                        const float* __restrict__ ssc, const float* __restrict__ qsc,
                        const float* __restrict__ g2, const float* __restrict__ b2,
                        const float* __restrict__ gsc, const float* __restrict__ bsc,
                        float* __restrict__ out, int n) {
  int o = blockIdx.x * 256 + threadIdx.x;
  if (o >= n) return;
  int ow = o % 112; int t = o / 112;
  int oh = t % 112; t /= 112;
  int co = t & 255;
  int b  = t >> 8;
  size_t e = (size_t)((b * 112 + oh) * 112 + ow) * 256 + co;
  float m2 = s2[co] * (1.0f / BN_COUNT);
  float v2 = q2[co] * (1.0f / BN_COUNT) - m2 * m2;
  float ms = ssc[co] * (1.0f / BN_COUNT);
  float vs = qsc[co] * (1.0f / BN_COUNT) - ms * ms;
  float a = (y2[e]  - m2) * rsqrtf(v2 + BN_EPS) * g2[co]  + b2[co];
  float c = (ysc[e] - ms) * rsqrtf(vs + BN_EPS) * gsc[co] + bsc[co];
  float r = a + c;
  out[o] = r > 0.f ? r : 0.f;
}

// ---------------------------------------------------------------- launch
extern "C" void kernel_launch(void* const* d_in, const int* in_sizes, int n_in,
                              void* d_out, int out_size, void* d_ws, size_t ws_size,
                              hipStream_t stream) {
  const float* x   = (const float*)d_in[0];
  const float* w1  = (const float*)d_in[1];
  const float* g1  = (const float*)d_in[2];
  const float* b1  = (const float*)d_in[3];
  const float* w2  = (const float*)d_in[4];
  const float* g2  = (const float*)d_in[5];
  const float* b2  = (const float*)d_in[6];
  const float* wsc = (const float*)d_in[7];
  const float* gsc = (const float*)d_in[8];
  const float* bsc = (const float*)d_in[9];

  char* ws = (char*)d_ws;
  size_t off = 0;
  auto take = [&](size_t bytes) -> char* {
    char* p = ws + off;
    off = (off + bytes + 255) & ~(size_t)255;
    return p;
  };

  float* stats = (float*)take(6 * 256 * sizeof(float));
  float* sum1 = stats;        float* sq1 = stats + 256;
  float* sumc = stats + 512;  float* sqc = stats + 768;
  float* sum2 = stats + 1024; float* sq2 = stats + 1280;

  unsigned short* xp   = (unsigned short*)take((size_t)8 * 56 * 56 * 512 * 2);
  unsigned short* w1p  = (unsigned short*)take((size_t)25 * 256 * 512 * 2);
  unsigned short* wscp = (unsigned short*)take((size_t)25 * 256 * 512 * 2);
  unsigned short* w2p  = (unsigned short*)take((size_t)9 * 256 * 256 * 2);
  float* y1  = (float*)take((size_t)8 * 256 * 112 * 112 * 4);
  float* ysc = (float*)take((size_t)8 * 256 * 112 * 112 * 4);
  unsigned short* a1p = (unsigned short*)take((size_t)8 * 112 * 112 * 256 * 2);
  float* y2  = (float*)take((size_t)8 * 256 * 112 * 112 * 4);

  k_zero<<<6, 256, 0, stream>>>(stats, 1536);
  k_pack_x<<<12845056 / 256, 256, 0, stream>>>(x, xp, 12845056);
  k_pack_w<<<3276800 / 256, 256, 0, stream>>>(w1,  w1p,  512, 5, 3276800);
  k_pack_w<<<3276800 / 256, 256, 0, stream>>>(wsc, wscp, 512, 5, 3276800);
  k_pack_w<<<589824 / 256, 256, 0, stream>>>(w2,  w2p,  256, 3, 589824);

  // conv1 / shortcut: zero-stuffed 5x5 decomposed into 4 parity phases.
  const int khE[3] = {0, 2, 4}, diE[3] = {-1, 0, 1};   // even output parity taps
  const int khO[2] = {1, 3},    diO[2] = {0, 1};       // odd output parity taps
  for (int ph = 0; ph < 2; ++ph) {
    for (int pw = 0; pw < 2; ++pw) {
      Taps tp; tp.n = 0;
      const int* khs = ph ? khO : khE; const int* dhs = ph ? diO : diE;
      const int  nh  = ph ? 2 : 3;
      const int* kws = pw ? khO : khE; const int* dws = pw ? diO : diE;
      const int  nw  = pw ? 2 : 3;
      for (int a = 0; a < nh; ++a)
        for (int c = 0; c < nw; ++c) {
          tp.di[tp.n]   = dhs[a];
          tp.dj[tp.n]   = dws[c];
          tp.woff[tp.n] = (khs[a] * 5 + kws[c]) * 256 * 512;
          tp.n++;
        }
      dim3 g(196, 2);   // M = 8*56*56 = 25088 -> 196 tiles of 128 ; N = 256 -> 2 tiles
      k_conv_wmma<<<g, 256, 0, stream>>>(xp, w1p,  y1,  sum1, sq1,
                                         56, 56, 512, 56, 56, 2, ph, 2, pw, tp);
      k_conv_wmma<<<g, 256, 0, stream>>>(xp, wscp, ysc, sumc, sqc,
                                         56, 56, 512, 56, 56, 2, ph, 2, pw, tp);
    }
  }

  k_bn_relu_pack<<<25690112 / 256, 256, 0, stream>>>(y1, sum1, sq1, g1, b1, a1p, 25690112);

  Taps t2; t2.n = 9;
  {
    int n = 0;
    for (int kh = 0; kh < 3; ++kh)
      for (int kw = 0; kw < 3; ++kw) {
        t2.di[n] = kh - 1;
        t2.dj[n] = kw - 1;
        t2.woff[n] = (kh * 3 + kw) * 256 * 256;
        n++;
      }
  }
  // M = 8*112*112 = 100352 -> 784 tiles of 128
  k_conv_wmma<<<dim3(784, 2), 256, 0, stream>>>(a1p, w2p, y2, sum2, sq2,
                                                112, 112, 256, 112, 112, 1, 0, 1, 0, t2);

  k_final<<<25690112 / 256, 256, 0, stream>>>(y2, ysc, sum2, sq2, sumc, sqc,
                                              g2, b2, gsc, bsc, (float*)d_out, 25690112);
}
